// MACE_65111704207442
// MI455X (gfx1250) — matrix operand order repaired and checked by reading, not compiled, not run on hardware
//
#include <hip/hip_runtime.h>
#include <hip/hip_bf16.h>

// ---------------------------------------------------------------------------
// MACE-like 2-layer message-passing energy model, forward + analytic backward
// (forces).  All dense GEMMs run through a bf16 WMMA (v_wmma_f32_16x16x32_bf16)
// tiled kernel; scatter/segment sums via global f32 atomics.
// ---------------------------------------------------------------------------

typedef __bf16 bf16;
typedef __attribute__((ext_vector_type(16))) __bf16 v16bf;
typedef __attribute__((ext_vector_type(8)))  float  v8f;

#define NN      3000
#define NPAD    3072
#define EE      96000
#define CC      128
#define NB_     8
#define NBP     32          // rf padded K (zero-filled 8..31)
#define NELEM   10
#define GG      6
#define RMAXF   5.0f
#define AVG_INV (1.0f / 32.0f)

constexpr int kLML[16] = {0,1,1,1,2,2,2,2,2,3,3,3,3,3,3,3};

__device__ __forceinline__ float siluf(float x) {
    float s = 1.0f / (1.0f + __expf(-x));
    return x * s;
}
__device__ __forceinline__ float dsiluf(float x) {
    float s = 1.0f / (1.0f + __expf(-x));
    return s * (1.0f + x * (1.0f - s));
}

// ---------------------------------------------------------------------------
// Generic tiled WMMA GEMM:  C[M,Nn] (+)= op(A[M,K]) @ B   (bf16 math, f32 acc)
//   B is ALWAYS given in [Nn][K] layout (row = output column), ldb = K-stride.
//   A_BF16 : A elements stored as bf16 (else f32)
//   ACT_A  : apply SiLU to A elements on load (A holds pre-activations)
//   OUT_BF16 / OUT_ADD : output dtype / accumulate into existing output
//   DSILU  : multiply result by silu'(Z[m,n]) (Z bf16) before store
// Block: 128 threads = 4 waves; tile 64(M) x 64(N); K-step 32.
// Fast-path staging assumes (and our shapes guarantee) M%64==0 and K%32==0.
// ---------------------------------------------------------------------------
template<bool A_BF16, bool ACT_A, bool OUT_BF16, bool OUT_ADD, bool DSILU>
__global__ void __launch_bounds__(128) mace_gemm_k(
    const void* __restrict__ Ap, int lda,
    const bf16* __restrict__ Bp, int ldb,
    void* __restrict__ Cp, int ldc,
    const bf16* __restrict__ Zp, int ldz,
    int M, int Nn, int K)
{
    __shared__ bf16 As[64 * 40];
    __shared__ bf16 Bs[64 * 40];
    const int tid  = threadIdx.x;
    const int wave = tid >> 5;
    const int lane = tid & 31;
    const int g    = lane >> 4;
    const int l15  = lane & 15;
    const int m0   = blockIdx.y * 64;
    const int n0   = blockIdx.x * 64;

    // staging coords: each thread fills one 16-wide chunk of one row
    const int rA  = tid >> 1;       // 0..63
    const int hfA = tid & 1;        // chunk (cols hf*16 .. +15)
    // A-fragment group->slot map: k-groups of 8: {0,1,2,3} -> slots {0,8,16,24} (contiguous)
    const int sb0 = (hfA == 0) ? 0 : 16;
    const int sb1 = (hfA == 0) ? 8 : 24;

    v8f acc[4];
#pragma unroll
    for (int nt = 0; nt < 4; ++nt)
#pragma unroll
        for (int j = 0; j < 8; ++j) acc[nt][j] = 0.0f;

    for (int k0 = 0; k0 < K; k0 += 32) {
        // ---------------- stage A tile ----------------
        {
            const int gm = m0 + rA;
            const int gk = k0 + hfA * 16;
            union { bf16 h[8]; uint4 q; } o0, o1;
            if (m0 + 64 <= M && k0 + 32 <= K) {
                if (A_BF16) {
                    const uint4* ap = (const uint4*)((const bf16*)Ap + (long)gm * lda + gk);
                    union { uint4 q; bf16 h[8]; } i0, i1;
                    i0.q = ap[0]; i1.q = ap[1];
                    if (ACT_A) {
#pragma unroll
                        for (int j = 0; j < 8; ++j) {
                            o0.h[j] = (bf16)siluf((float)i0.h[j]);
                            o1.h[j] = (bf16)siluf((float)i1.h[j]);
                        }
                    } else { o0.q = i0.q; o1.q = i1.q; }
                } else {
                    const float4* ap = (const float4*)((const float*)Ap + (long)gm * lda + gk);
                    float4 f0 = ap[0], f1 = ap[1], f2 = ap[2], f3 = ap[3];
                    float v[16] = {f0.x, f0.y, f0.z, f0.w, f1.x, f1.y, f1.z, f1.w,
                                   f2.x, f2.y, f2.z, f2.w, f3.x, f3.y, f3.z, f3.w};
#pragma unroll
                    for (int j = 0; j < 8; ++j) {
                        float a = v[j], b = v[8 + j];
                        if (ACT_A) { a = siluf(a); b = siluf(b); }
                        o0.h[j] = (bf16)a;
                        o1.h[j] = (bf16)b;
                    }
                }
            } else {  // guarded fallback (not hit for our shapes)
#pragma unroll
                for (int j = 0; j < 16; ++j) {
                    float v = 0.0f;
                    int kk = gk + j;
                    if (gm < M && kk < K) {
                        if (A_BF16) v = (float)((const bf16*)Ap)[(long)gm * lda + kk];
                        else        v = ((const float*)Ap)[(long)gm * lda + kk];
                        if (ACT_A) v = siluf(v);
                    }
                    if (j < 8) o0.h[j] = (bf16)v; else o1.h[j - 8] = (bf16)v;
                }
            }
            *(uint4*)&As[rA * 40 + sb0] = o0.q;
            *(uint4*)&As[rA * 40 + sb1] = o1.q;
        }
        // ---------------- stage B tile ([n][k] layout; contiguous) ----------------
        {
            const int gn = n0 + rA;
            const int gk = k0 + hfA * 16;
            union { bf16 h[8]; uint4 q; } b0, b1;
            if (gn < Nn && gk + 16 <= K) {
                const uint4* bp = (const uint4*)(Bp + (long)gn * ldb + gk);
                b0.q = bp[0]; b1.q = bp[1];
            } else if (gn < Nn) {
#pragma unroll
                for (int j = 0; j < 16; ++j) {
                    int kk = gk + j;
                    bf16 v = (bf16)0.0f;
                    if (kk < K) v = Bp[(long)gn * ldb + kk];
                    if (j < 8) b0.h[j] = v; else b1.h[j - 8] = v;
                }
            } else {
#pragma unroll
                for (int j = 0; j < 8; ++j) { b0.h[j] = (bf16)0.0f; b1.h[j] = (bf16)0.0f; }
            }
            *(uint4*)&Bs[rA * 40 + hfA * 16]     = b0.q;
            *(uint4*)&Bs[rA * 40 + hfA * 16 + 8] = b1.q;
        }
        __syncthreads();

        union { uint4 u[2]; v16bf v; } af, bf;
        {
            const uint4* pa = (const uint4*)&As[(wave * 16 + l15) * 40 + g * 16];
            af.u[0] = pa[0]; af.u[1] = pa[1];
        }
#pragma unroll
        for (int nt = 0; nt < 4; ++nt) {
            const uint4* pb = (const uint4*)&Bs[(nt * 16 + l15) * 40 + g * 16];
            bf.u[0] = pb[0]; bf.u[1] = pb[1];
            acc[nt] = __builtin_amdgcn_wmma_f32_16x16x32_bf16(
                false, af.v, false, bf.v, (short)0, acc[nt], false, false);
        }
        __syncthreads();
    }

    // ---- epilogue ----
#pragma unroll
    for (int nt = 0; nt < 4; ++nt) {
#pragma unroll
        for (int j = 0; j < 8; ++j) {
            int m = m0 + wave * 16 + j + 8 * g;
            int n = n0 + nt * 16 + l15;
            if (m < M && n < Nn) {
                float v = acc[nt][j];
                if (DSILU) v *= dsiluf((float)Zp[(long)m * ldz + n]);
                if (OUT_BF16) {
                    bf16* o = (bf16*)Cp + (long)m * ldc + n;
                    if (OUT_ADD) v += (float)(*o);
                    *o = (bf16)v;
                } else {
                    float* o = (float*)Cp + (long)m * ldc + n;
                    if (OUT_ADD) v += *o;
                    *o = v;
                }
            }
        }
    }
}

template<bool A_BF16, bool ACT_A, bool OUT_BF16, bool OUT_ADD, bool DSILU>
static void launch_gemm(hipStream_t st, const void* A, int lda, const bf16* B, int ldb,
                        void* C, int ldc, const bf16* Z, int ldz, int M, int Nn, int K)
{
    dim3 grid((Nn + 63) / 64, (M + 63) / 64);
    mace_gemm_k<A_BF16, ACT_A, OUT_BF16, OUT_ADD, DSILU>
        <<<grid, 128, 0, st>>>(A, lda, B, ldb, C, ldc, Z, ldz, M, Nn, K);
}

// ---------------------------------------------------------------------------
// Weight conversion: plain copy and transpose (f32 -> bf16)
// ---------------------------------------------------------------------------
__global__ void mace_f2bf_k(const float* __restrict__ s, bf16* __restrict__ d, int n)
{
    int i = blockIdx.x * 256 + threadIdx.x;
    if (i < n) d[i] = (bf16)s[i];
}

// dT[n*ldk + k] = s[k*N + n]   (src is [K][N])
__global__ void mace_f2bfT_k(const float* __restrict__ s, bf16* __restrict__ d,
                             int K, int N, int ldk)
{
    int i = blockIdx.x * 256 + threadIdx.x;
    if (i >= K * N) return;
    int k = i / N, n = i - k * N;
    d[(long)n * ldk + k] = (bf16)s[i];
}

// ---------------------------------------------------------------------------
// Edge geometry: vec, r, 1/r, Y(16), rf (padded stride 32)
// ---------------------------------------------------------------------------
__global__ void mace_edge_geom_k(const float* __restrict__ pos, const float* __restrict__ shifts,
                                 const int* __restrict__ eidx,
                                 float* __restrict__ vecb, float* __restrict__ rb,
                                 float* __restrict__ rib, float* __restrict__ Yb,
                                 float* __restrict__ rfb)
{
    int e = blockIdx.x * 256 + threadIdx.x;
    if (e >= EE) return;
    int s = eidx[e], rv = eidx[EE + e];
    float vx = pos[rv * 3 + 0] - pos[s * 3 + 0] + shifts[e * 3 + 0];
    float vy = pos[rv * 3 + 1] - pos[s * 3 + 1] + shifts[e * 3 + 1];
    float vz = pos[rv * 3 + 2] - pos[s * 3 + 2] + shifts[e * 3 + 2];
    float r2 = vx * vx + vy * vy + vz * vz;
    float r  = sqrtf(fmaxf(r2, 1e-24f));
    float ri = 1.0f / r;
    vecb[e * 3 + 0] = vx; vecb[e * 3 + 1] = vy; vecb[e * 3 + 2] = vz;
    rb[e] = r; rib[e] = ri;
    float x = vx * ri, y = vy * ri, z = vz * ri;
    const float s3 = 1.7320508075688772f, s15 = 3.872983346207417f;
    const float s5h = 1.118033988749895f, s15h = 1.9364916731037085f;
    const float c9 = 2.091650066335189f, c10 = 10.246950765959598f;
    const float c11 = 1.6201851746019651f, c12h = 1.3228756555322954f;
    const float c14 = 5.123475382979799f;
    float* Y = Yb + e * 16;
    Y[0]  = 1.0f;           Y[1]  = s3 * x;          Y[2]  = s3 * y;        Y[3] = s3 * z;
    Y[4]  = s15 * x * y;    Y[5]  = s15 * y * z;     Y[6]  = s5h * (3.0f * z * z - 1.0f);
    Y[7]  = s15 * x * z;    Y[8]  = s15h * (x * x - y * y);
    Y[9]  = c9 * y * (3.0f * x * x - y * y);
    Y[10] = c10 * x * y * z;
    Y[11] = c11 * y * (5.0f * z * z - 1.0f);
    Y[12] = c12h * (5.0f * z * z * z - 3.0f * z);
    Y[13] = c11 * x * (5.0f * z * z - 1.0f);
    Y[14] = c14 * z * (x * x - y * y);
    Y[15] = c9 * x * (3.0f * x * x - y * y);
    float u = r * (1.0f / RMAXF);
    float u2 = u * u, u4 = u2 * u2, u5 = u4 * u, u6 = u5 * u, u7 = u6 * u;
    float fe = (u < 1.0f) ? (1.0f - 21.0f * u5 + 35.0f * u6 - 15.0f * u7) : 0.0f;
    const float sq2r = 0.6324555320336759f;   // sqrt(2/5)
    const float pir  = 0.6283185307179586f;   // pi/5
#pragma unroll
    for (int n = 1; n <= 8; ++n)
        rfb[(long)e * NBP + n - 1] = sq2r * __sinf((float)n * pir * r) * ri * fe;
}

// ---------------------------------------------------------------------------
// Node init: species, h0 = W_embed[z], node_e0, total
// ---------------------------------------------------------------------------
__global__ void __launch_bounds__(128) mace_node_init_k(
    const float* __restrict__ attrs, const float* __restrict__ Wemb,
    const float* __restrict__ ae, const int* __restrict__ batch,
    float* __restrict__ h0, int* __restrict__ spec,
    float* __restrict__ outTotal, float* __restrict__ outNodeE)
{
    int n = blockIdx.x;
    __shared__ int zs;
    if (threadIdx.x == 0) {
        int z = 0;
        for (int j = 0; j < NELEM; ++j)
            if (attrs[n * NELEM + j] > 0.5f) z = j;
        zs = z; spec[n] = z;
        float e0 = ae[z];
        outNodeE[n] = e0;
        atomicAdd(&outTotal[batch[n]], e0);
    }
    __syncthreads();
    h0[n * CC + threadIdx.x] = Wemb[zs * CC + threadIdx.x];
}

// ---------------------------------------------------------------------------
// Forward messages: Araw[recv,lm,c] += w[e,L(lm),c]*Y[e,lm]*hu[send,c]/AVG
// ---------------------------------------------------------------------------
__global__ void __launch_bounds__(128) mace_msg_fwd_k(
    const bf16* __restrict__ wb, const float* __restrict__ Yb,
    const float* __restrict__ hu, const int* __restrict__ eidx,
    float* __restrict__ Araw)
{
    int e = blockIdx.x;
    __shared__ float Ys[16];
    __shared__ int ss[2];
    if (threadIdx.x < 16) Ys[threadIdx.x] = Yb[e * 16 + threadIdx.x];
    if (threadIdx.x == 0) { ss[0] = eidx[e]; ss[1] = eidx[EE + e]; }
    __syncthreads();
    int c = threadIdx.x;
    float hus = hu[ss[0] * CC + c];
    const bf16* wr = wb + (long)e * 4 * CC + c;
    float wl[4];
#pragma unroll
    for (int l = 0; l < 4; ++l) wl[l] = (float)wr[l * CC];
    float* ar = Araw + (long)ss[1] * 16 * CC + c;
#pragma unroll
    for (int lm = 0; lm < 16; ++lm)
        atomicAdd(&ar[lm * CC], wl[kLML[lm]] * Ys[lm] * hus * AVG_INV);
}

// ---------------------------------------------------------------------------
// sc[n,k] = sum_c h[n,c] * Wsc[c, z_n, k]
// ---------------------------------------------------------------------------
__global__ void __launch_bounds__(128) mace_sc_fwd_k(
    const float* __restrict__ h, const float* __restrict__ Wsc,
    const int* __restrict__ spec, float* __restrict__ hout)
{
    int n = blockIdx.x;
    __shared__ float hs[CC];
    hs[threadIdx.x] = h[n * CC + threadIdx.x];
    __syncthreads();
    int z = spec[n], k = threadIdx.x;
    float acc = 0.0f;
    for (int c = 0; c < CC; ++c)
        acc += hs[c] * Wsc[((long)c * NELEM + z) * CC + k];
    hout[n * CC + k] = acc;
}

// ---------------------------------------------------------------------------
// basis + mix:  mix[n,c] = sum_p Wcontr[z,p,c]*basis[p]
// ---------------------------------------------------------------------------
__global__ void __launch_bounds__(128) mace_basis_mix_k(
    const float* __restrict__ A, const float* __restrict__ Wcontr,
    const int* __restrict__ spec, float* __restrict__ mix)
{
    int n = blockIdx.x, c = threadIdx.x;
    const float* a = A + (long)n * 16 * CC + c;
    float a0 = a[0];
    float nl0 = a0 * a0, nl1 = 0.0f, nl2 = 0.0f, nl3 = 0.0f;
#pragma unroll
    for (int lm = 1; lm < 4; ++lm) { float v = a[lm * CC]; nl1 += v * v; }
#pragma unroll
    for (int lm = 4; lm < 9; ++lm) { float v = a[lm * CC]; nl2 += v * v; }
#pragma unroll
    for (int lm = 9; lm < 16; ++lm) { float v = a[lm * CC]; nl3 += v * v; }
    int z = spec[n];
    const float* Wc = Wcontr + (long)z * 9 * CC + c;
    float m = Wc[0 * CC] * a0 + Wc[1 * CC] * nl0 + Wc[2 * CC] * nl1
            + Wc[3 * CC] * nl2 + Wc[4 * CC] * nl3
            + Wc[5 * CC] * (a0 * nl0) + Wc[6 * CC] * (a0 * nl1)
            + Wc[7 * CC] * (a0 * nl2) + Wc[8 * CC] * (a0 * nl3);
    mix[n * CC + c] = m;
}

// ---------------------------------------------------------------------------
// Readouts
// ---------------------------------------------------------------------------
__global__ void __launch_bounds__(128) mace_readout0_k(
    const float* __restrict__ h, const float* __restrict__ wr0,
    const int* __restrict__ batch, float* __restrict__ outTotal,
    float* __restrict__ outNodeE)
{
    int n = blockIdx.x;
    __shared__ float red[128];
    red[threadIdx.x] = h[n * CC + threadIdx.x] * wr0[threadIdx.x];
    __syncthreads();
    for (int s = 64; s > 0; s >>= 1) {
        if (threadIdx.x < s) red[threadIdx.x] += red[threadIdx.x + s];
        __syncthreads();
    }
    if (threadIdx.x == 0) {
        float ne = red[0];
        outNodeE[n] += ne;
        atomicAdd(&outTotal[batch[n]], ne);
    }
}

__global__ void __launch_bounds__(128) mace_readout1_k(
    const float* __restrict__ h, const float* __restrict__ wr1a,
    const float* __restrict__ wr1b, const int* __restrict__ batch,
    float* __restrict__ tb, float* __restrict__ outTotal, float* __restrict__ outNodeE)
{
    int n = blockIdx.x;
    __shared__ float hs[CC];
    __shared__ float ts[16];
    hs[threadIdx.x] = h[n * CC + threadIdx.x];
    __syncthreads();
    if (threadIdx.x < 16) {
        int j = threadIdx.x;
        float t = 0.0f;
        for (int c = 0; c < CC; ++c) t += hs[c] * wr1a[c * 16 + j];
        tb[n * 16 + j] = t;
        ts[j] = siluf(t) * wr1b[j];
    }
    __syncthreads();
    if (threadIdx.x == 0) {
        float ne = 0.0f;
#pragma unroll
        for (int j = 0; j < 16; ++j) ne += ts[j];
        outNodeE[n] += ne;
        atomicAdd(&outTotal[batch[n]], ne);
    }
}

// ---------------------------------------------------------------------------
// Backward pieces
// ---------------------------------------------------------------------------
__global__ void __launch_bounds__(128) mace_grad_readout1_k(
    const float* __restrict__ tb, const float* __restrict__ wr1a,
    const float* __restrict__ wr1b, float* __restrict__ gh)
{
    int n = blockIdx.x;
    __shared__ float ds[16];
    if (threadIdx.x < 16)
        ds[threadIdx.x] = dsiluf(tb[n * 16 + threadIdx.x]) * wr1b[threadIdx.x];
    __syncthreads();
    int c = threadIdx.x;
    float g = 0.0f;
#pragma unroll
    for (int j = 0; j < 16; ++j) g += ds[j] * wr1a[c * 16 + j];
    gh[n * CC + c] = g;
}

__global__ void __launch_bounds__(128) mace_sc_bwd_k(
    const float* __restrict__ gout, const float* __restrict__ Wsc,
    const int* __restrict__ spec, float* __restrict__ gh)
{
    int n = blockIdx.x;
    __shared__ float gs[CC];
    gs[threadIdx.x] = gout[n * CC + threadIdx.x];
    __syncthreads();
    int z = spec[n], c = threadIdx.x;
    float acc = 0.0f;
    for (int k = 0; k < CC; ++k)
        acc += gs[k] * Wsc[((long)c * NELEM + z) * CC + k];
    gh[n * CC + c] = acc;
}

__global__ void __launch_bounds__(128) mace_basis_bwd_k(
    const float* __restrict__ A, const float* __restrict__ Wcontr,
    const int* __restrict__ spec, const float* __restrict__ gmix,
    float* __restrict__ gA)
{
    int n = blockIdx.x, c = threadIdx.x;
    const float* a = A + (long)n * 16 * CC + c;
    float a0 = a[0];
    float nl[4];
    nl[0] = a0 * a0; nl[1] = 0.0f; nl[2] = 0.0f; nl[3] = 0.0f;
#pragma unroll
    for (int lm = 1; lm < 4; ++lm) { float v = a[lm * CC]; nl[1] += v * v; }
#pragma unroll
    for (int lm = 4; lm < 9; ++lm) { float v = a[lm * CC]; nl[2] += v * v; }
#pragma unroll
    for (int lm = 9; lm < 16; ++lm) { float v = a[lm * CC]; nl[3] += v * v; }
    int z = spec[n];
    float gm = gmix[n * CC + c];
    const float* Wc = Wcontr + (long)z * 9 * CC + c;
    float gb[9];
#pragma unroll
    for (int p = 0; p < 9; ++p) gb[p] = Wc[p * CC] * gm;
    float ga0 = gb[0] + gb[5] * nl[0] + gb[6] * nl[1] + gb[7] * nl[2] + gb[8] * nl[3];
    float gnl[4];
#pragma unroll
    for (int l = 0; l < 4; ++l) gnl[l] = gb[1 + l] + gb[5 + l] * a0;
    float* go = gA + (long)n * 16 * CC + c;
#pragma unroll
    for (int lm = 0; lm < 16; ++lm) {
        float v = 2.0f * a[lm * CC] * gnl[kLML[lm]];
        if (lm == 0) v += ga0;
        go[lm * CC] = v;
    }
}

// Backward messages: consumes w (bf16), overwrites it with g_w; accumulates
// g_Y (per-edge) and scatters g_hu (atomics).
__global__ void __launch_bounds__(128) mace_msg_bwd_k(
    bf16* __restrict__ wb, const float* __restrict__ Yb,
    const float* __restrict__ hu, const int* __restrict__ eidx,
    const float* __restrict__ gS, float* __restrict__ gY, float* __restrict__ ghu)
{
    int e = blockIdx.x;
    __shared__ float Ys[16];
    __shared__ float gYs[16];
    __shared__ int ss[2];
    if (threadIdx.x < 16) { Ys[threadIdx.x] = Yb[e * 16 + threadIdx.x]; gYs[threadIdx.x] = 0.0f; }
    if (threadIdx.x == 0) { ss[0] = eidx[e]; ss[1] = eidx[EE + e]; }
    __syncthreads();
    int c = threadIdx.x;
    float hus = hu[ss[0] * CC + c];
    bf16* wr = wb + (long)e * 4 * CC + c;
    float wl[4];
#pragma unroll
    for (int l = 0; l < 4; ++l) wl[l] = (float)wr[l * CC];
    const float* gs = gS + (long)ss[1] * 16 * CC + c;
    float gw[4] = {0.0f, 0.0f, 0.0f, 0.0f};
    float gh = 0.0f;
#pragma unroll
    for (int lm = 0; lm < 16; ++lm) {
        int l = kLML[lm];
        float gm = gs[lm * CC] * AVG_INV;
        gw[l] += gm * Ys[lm] * hus;
        gh += gm * Ys[lm] * wl[l];
        atomicAdd(&gYs[lm], gm * wl[l] * hus);
    }
#pragma unroll
    for (int l = 0; l < 4; ++l) wr[l * CC] = (bf16)gw[l];
    atomicAdd(&ghu[ss[0] * CC + c], gh);
    __syncthreads();
    if (threadIdx.x < 16) gY[e * 16 + threadIdx.x] += gYs[threadIdx.x];
}

__global__ void __launch_bounds__(128) mace_add_r0_k(
    const float* __restrict__ gh, const float* __restrict__ wr0,
    float* __restrict__ ghcur)
{
    int n = blockIdx.x;
    ghcur[n * CC + threadIdx.x] = gh[n * CC + threadIdx.x] + wr0[threadIdx.x];
}

// ---------------------------------------------------------------------------
// Edge geometry backward: g_Y, g_rf -> g_vec -> force scatter
// ---------------------------------------------------------------------------
__global__ void mace_geom_bwd_k(
    const float* __restrict__ vecb, const float* __restrict__ rb,
    const float* __restrict__ rib, const float* __restrict__ gY,
    const float* __restrict__ grf, const int* __restrict__ eidx,
    float* __restrict__ forces)
{
    int e = blockIdx.x * 256 + threadIdx.x;
    if (e >= EE) return;
    float r = rb[e], ri = rib[e];
    float x = vecb[e * 3 + 0] * ri, y = vecb[e * 3 + 1] * ri, z = vecb[e * 3 + 2] * ri;
    const float* gy = gY + e * 16;
    const float s3 = 1.7320508075688772f, s15 = 3.872983346207417f;
    const float s5h = 1.118033988749895f;
    const float c9 = 2.091650066335189f, c10 = 10.246950765959598f;
    const float c11 = 1.6201851746019651f, c12h = 1.3228756555322954f;
    const float c14 = 5.123475382979799f;
    float gux = gy[1] * s3 + gy[4] * s15 * y + gy[7] * s15 * z + gy[8] * s15 * x
              + gy[9] * c9 * 6.0f * x * y + gy[10] * c10 * y * z
              + gy[13] * c11 * (5.0f * z * z - 1.0f)
              + gy[14] * c14 * 2.0f * x * z + gy[15] * c9 * (9.0f * x * x - y * y);
    float guy = gy[2] * s3 + gy[4] * s15 * x + gy[5] * s15 * z - gy[8] * s15 * y
              + gy[9] * c9 * (3.0f * x * x - 3.0f * y * y) + gy[10] * c10 * x * z
              + gy[11] * c11 * (5.0f * z * z - 1.0f)
              - gy[14] * c14 * 2.0f * y * z - gy[15] * c9 * 2.0f * x * y;
    float guz = gy[3] * s3 + gy[5] * s15 * y + gy[6] * s5h * 6.0f * z + gy[7] * s15 * x
              + gy[10] * c10 * x * y + gy[11] * c11 * 10.0f * y * z
              + gy[12] * c12h * (15.0f * z * z - 3.0f)
              + gy[13] * c11 * 10.0f * x * z + gy[14] * c14 * (x * x - y * y);
    float dot = x * gux + y * guy + z * guz;
    float gvx = (gux - x * dot) * ri;
    float gvy = (guy - y * dot) * ri;
    float gvz = (guz - z * dot) * ri;
    float u = r * (1.0f / RMAXF);
    float gr = 0.0f;
    if (u < 1.0f) {
        float u2 = u * u, u4 = u2 * u2, u5 = u4 * u, u6 = u5 * u;
        float fe = 1.0f - 21.0f * u5 + 35.0f * u6 - 15.0f * u6 * u;
        float fp = (-105.0f * u4 + 210.0f * u5 - 105.0f * u6) * (1.0f / RMAXF);
        const float sq2r = 0.6324555320336759f;
        const float pir  = 0.6283185307179586f;
#pragma unroll
        for (int nb = 1; nb <= 8; ++nb) {
            float bn = (float)nb * pir;
            float sn = __sinf(bn * r), cn = __cosf(bn * r);
            float drf = sq2r * ((bn * cn * ri - sn * ri * ri) * fe + sn * ri * fp);
            gr += grf[e * 8 + nb - 1] * drf;
        }
    }
    gvx += gr * x; gvy += gr * y; gvz += gr * z;
    int s = eidx[e], rv = eidx[EE + e];
    atomicAdd(&forces[rv * 3 + 0], -gvx);
    atomicAdd(&forces[rv * 3 + 1], -gvy);
    atomicAdd(&forces[rv * 3 + 2], -gvz);
    atomicAdd(&forces[s * 3 + 0], gvx);
    atomicAdd(&forces[s * 3 + 1], gvy);
    atomicAdd(&forces[s * 3 + 2], gvz);
}

// ---------------------------------------------------------------------------
// Host orchestration
// ---------------------------------------------------------------------------
extern "C" void kernel_launch(void* const* d_in, const int* in_sizes, int n_in,
                              void* d_out, int out_size, void* d_ws, size_t ws_size,
                              hipStream_t stream)
{
    (void)in_sizes; (void)n_in;
    const float* positions  = (const float*)d_in[0];
    const float* node_attrs = (const float*)d_in[1];
    const float* shifts     = (const float*)d_in[2];
    const int*   eidx       = (const int*)d_in[3];
    const int*   batch      = (const int*)d_in[4];
    const float* atomic_e   = (const float*)d_in[5];
    const float* W_embed    = (const float*)d_in[6];
    const float* wr0        = (const float*)d_in[7];
    const float* wr1a       = (const float*)d_in[8];
    const float* wr1b       = (const float*)d_in[9];
    const float* rm[2][4]; const float* Wup[2]; const float* Wpost[2];
    const float* Wsc[2]; const float* Wcontr[2]; const float* Wout[2];
    for (int i = 0; i < 2; ++i) {
        int b = 10 + i * 9;
        rm[i][0] = (const float*)d_in[b + 0];
        rm[i][1] = (const float*)d_in[b + 1];
        rm[i][2] = (const float*)d_in[b + 2];
        rm[i][3] = (const float*)d_in[b + 3];
        Wup[i]    = (const float*)d_in[b + 4];
        Wpost[i]  = (const float*)d_in[b + 5];
        Wsc[i]    = (const float*)d_in[b + 6];
        Wcontr[i] = (const float*)d_in[b + 7];
        Wout[i]   = (const float*)d_in[b + 8];
    }

    float* outTotal = (float*)d_out;            // [6]
    float* outNodeE = (float*)d_out + GG;       // [3000]
    float* forces   = (float*)d_out + GG + NN;  // [3000*3]

    // ---- workspace carve ----
    char* wp = (char*)d_ws;
    auto alloc = [&](size_t bytes) -> void* {
        void* r = (void*)wp;
        wp += (bytes + 255) & ~(size_t)255;
        return r;
    };
    float* vecb  = (float*)alloc((size_t)EE * 3 * 4);
    float* rbuf  = (float*)alloc((size_t)EE * 4);
    float* ribuf = (float*)alloc((size_t)EE * 4);
    float* Ybuf  = (float*)alloc((size_t)EE * 16 * 4);
    float* rfbuf = (float*)alloc((size_t)EE * NBP * 4);  // padded, cols 8..31 zero
    float* h0    = (float*)alloc((size_t)NPAD * CC * 4);
    float* h1    = (float*)alloc((size_t)NPAD * CC * 4);
    float* h2    = (float*)alloc((size_t)NPAD * CC * 4);
    float* hu0   = (float*)alloc((size_t)NPAD * CC * 4);
    float* hu1   = (float*)alloc((size_t)NPAD * CC * 4);
    float* A0    = (float*)alloc((size_t)NPAD * 16 * CC * 4);
    float* A1    = (float*)alloc((size_t)NPAD * 16 * CC * 4);
    float* Araw  = (float*)alloc((size_t)NPAD * 16 * CC * 4); // fwd Araw / bwd g_S
    float* gAbuf = (float*)alloc((size_t)NPAD * 16 * CC * 4);
    float* gmix  = (float*)alloc((size_t)NPAD * CC * 4);
    float* g_h   = (float*)alloc((size_t)NPAD * CC * 4);
    float* ghcur = (float*)alloc((size_t)NPAD * CC * 4);
    float* ghu   = (float*)alloc((size_t)NPAD * CC * 4);
    float* mixb  = (float*)alloc((size_t)NPAD * CC * 4);
    float* tbuf  = (float*)alloc((size_t)NPAD * 16 * 4);
    int*   spec  = (int*)alloc((size_t)NPAD * 4);
    float* gY    = (float*)alloc((size_t)EE * 16 * 4);
    float* grf   = (float*)alloc((size_t)EE * 8 * 4);
    bf16* z1   = (bf16*)alloc((size_t)EE * 64 * 2);
    bf16* z2   = (bf16*)alloc((size_t)EE * 64 * 2);
    bf16* z3   = (bf16*)alloc((size_t)EE * 64 * 2);
    bf16* gza  = (bf16*)alloc((size_t)EE * 64 * 2);
    bf16* gzb  = (bf16*)alloc((size_t)EE * 64 * 2);
    bf16* wbuf = (bf16*)alloc((size_t)EE * 4 * CC * 2);   // fwd w / bwd g_w
    // bf16 weights: forward uses transposed [N][K]; backward uses original [K][N]
    bf16 *WT1[2], *WT2[2], *WT3[2], *WT4[2], *WTup[2], *WTpost[2], *WTout[2];
    bf16 *W1o[2], *W2o[2], *W3o[2], *W4o[2], *Wupo[2], *Wposto[2], *Wouto[2];
    for (int i = 0; i < 2; ++i) {
        WT1[i]    = (bf16*)alloc(64 * NBP * 2);      // zero-padded K 8->32
        WT2[i]    = (bf16*)alloc(64 * 64 * 2);
        WT3[i]    = (bf16*)alloc(64 * 64 * 2);
        WT4[i]    = (bf16*)alloc(512 * 64 * 2);
        WTup[i]   = (bf16*)alloc(CC * CC * 2);
        WTpost[i] = (bf16*)alloc(4 * CC * CC * 2);
        WTout[i]  = (bf16*)alloc(CC * CC * 2);
        W1o[i]    = (bf16*)alloc(8 * 64 * 2);
        W2o[i]    = (bf16*)alloc(64 * 64 * 2);
        W3o[i]    = (bf16*)alloc(64 * 64 * 2);
        W4o[i]    = (bf16*)alloc(64 * 512 * 2);
        Wupo[i]   = (bf16*)alloc(CC * CC * 2);
        Wposto[i] = (bf16*)alloc(4 * CC * CC * 2);
        Wouto[i]  = (bf16*)alloc(CC * CC * 2);
    }
    size_t used = (size_t)(wp - (char*)d_ws);
    if (used > ws_size) used = ws_size;

    // ---- zero accumulators / padding ----
    hipMemsetAsync(d_out, 0, (size_t)out_size * sizeof(float), stream);
    hipMemsetAsync(d_ws, 0, used, stream);

    // ---- weight conversion to bf16 (both layouts) ----
    auto conv = [&](const float* s, bf16* d, int n) {
        mace_f2bf_k<<<(n + 255) / 256, 256, 0, stream>>>(s, d, n);
    };
    auto convT = [&](const float* s, bf16* d, int K, int N, int ldk) {
        mace_f2bfT_k<<<(K * N + 255) / 256, 256, 0, stream>>>(s, d, K, N, ldk);
    };
    for (int i = 0; i < 2; ++i) {
        convT(rm[i][0], WT1[i], 8, 64, NBP);
        convT(rm[i][1], WT2[i], 64, 64, 64);
        convT(rm[i][2], WT3[i], 64, 64, 64);
        convT(rm[i][3], WT4[i], 64, 512, 64);
        convT(Wup[i], WTup[i], CC, CC, CC);
        for (int l = 0; l < 4; ++l)
            convT(Wpost[i] + (size_t)l * CC * CC, WTpost[i] + (size_t)l * CC * CC, CC, CC, CC);
        convT(Wout[i], WTout[i], CC, CC, CC);
        conv(rm[i][0], W1o[i], 8 * 64);
        conv(rm[i][1], W2o[i], 64 * 64);
        conv(rm[i][2], W3o[i], 64 * 64);
        conv(rm[i][3], W4o[i], 64 * 512);
        conv(Wup[i], Wupo[i], CC * CC);
        conv(Wpost[i], Wposto[i], 4 * CC * CC);
        conv(Wout[i], Wouto[i], CC * CC);
    }

    // ---- forward ----
    mace_edge_geom_k<<<(EE + 255) / 256, 256, 0, stream>>>(
        positions, shifts, eidx, vecb, rbuf, ribuf, Ybuf, rfbuf);
    mace_node_init_k<<<NN, 128, 0, stream>>>(
        node_attrs, W_embed, atomic_e, batch, h0, spec, outTotal, outNodeE);

    float* hb[3] = {h0, h1, h2};
    for (int i = 0; i < 2; ++i) {
        float* hin  = hb[i];
        float* hout = hb[i + 1];
        float* hu   = i ? hu1 : hu0;
        float* Ai   = i ? A1 : A0;
        if (i == 1)
            hipMemsetAsync(Araw, 0, (size_t)NPAD * 16 * CC * 4, stream);
        // hu = h @ Wup
        launch_gemm<false,false,false,false,false>(stream, hin, CC, WTup[i], CC,
                                                   hu, CC, nullptr, 0, NPAD, CC, CC);
        // radial MLP (pre-activations stored; SiLU fused on next A-load)
        launch_gemm<false,false,true,false,false>(stream, rfbuf, NBP, WT1[i], NBP,
                                                  z1, 64, nullptr, 0, EE, 64, NBP);
        launch_gemm<true,true,true,false,false>(stream, z1, 64, WT2[i], 64,
                                                z2, 64, nullptr, 0, EE, 64, 64);
        launch_gemm<true,true,true,false,false>(stream, z2, 64, WT3[i], 64,
                                                z3, 64, nullptr, 0, EE, 64, 64);
        launch_gemm<true,true,true,false,false>(stream, z3, 64, WT4[i], 64,
                                                wbuf, 512, nullptr, 0, EE, 512, 64);
        // messages -> Araw (segment sum, 1/AVG folded in)
        mace_msg_fwd_k<<<EE, 128, 0, stream>>>(wbuf, Ybuf, hu, eidx, Araw);
        // per-lm W_post
        for (int lm = 0; lm < 16; ++lm) {
            int l = kLML[lm];
            launch_gemm<false,false,false,false,false>(
                stream, Araw + lm * CC, 16 * CC, WTpost[i] + (size_t)l * CC * CC, CC,
                Ai + lm * CC, 16 * CC, nullptr, 0, NPAD, CC, CC);
        }
        // sc + basis/mix + h_out = mix@Wout + sc
        mace_sc_fwd_k<<<NN, 128, 0, stream>>>(hin, Wsc[i], spec, hout);
        mace_basis_mix_k<<<NN, 128, 0, stream>>>(Ai, Wcontr[i], spec, mixb);
        launch_gemm<false,false,false,true,false>(stream, mixb, CC, WTout[i], CC,
                                                  hout, CC, nullptr, 0, NPAD, CC, CC);
        if (i == 0)
            mace_readout0_k<<<NN, 128, 0, stream>>>(hout, wr0, batch, outTotal, outNodeE);
        else
            mace_readout1_k<<<NN, 128, 0, stream>>>(hout, wr1a, wr1b, batch, tbuf,
                                                    outTotal, outNodeE);
    }

    // ---- backward (forces) ----
    mace_grad_readout1_k<<<NN, 128, 0, stream>>>(tbuf, wr1a, wr1b, ghcur);
    for (int i = 1; i >= 0; --i) {
        float* hu = i ? hu1 : hu0;
        float* Ai = i ? A1 : A0;
        if (i == 0) {  // recompute MLP intermediates for layer 0
            launch_gemm<false,false,true,false,false>(stream, rfbuf, NBP, WT1[0], NBP,
                                                      z1, 64, nullptr, 0, EE, 64, NBP);
            launch_gemm<true,true,true,false,false>(stream, z1, 64, WT2[0], 64,
                                                    z2, 64, nullptr, 0, EE, 64, 64);
            launch_gemm<true,true,true,false,false>(stream, z2, 64, WT3[0], 64,
                                                    z3, 64, nullptr, 0, EE, 64, 64);
            launch_gemm<true,true,true,false,false>(stream, z3, 64, WT4[0], 64,
                                                    wbuf, 512, nullptr, 0, EE, 512, 64);
        }
        // g_mix = g_out @ Wout^T ; g_h = sc-backprop(g_out)
        launch_gemm<false,false,false,false,false>(stream, ghcur, CC, Wouto[i], CC,
                                                   gmix, CC, nullptr, 0, NPAD, CC, CC);
        mace_sc_bwd_k<<<NN, 128, 0, stream>>>(ghcur, Wsc[i], spec, g_h);
        mace_basis_bwd_k<<<NN, 128, 0, stream>>>(Ai, Wcontr[i], spec, gmix, gAbuf);
        // g_S = gA @ Wpost^T (per lm) -> Araw buffer reused as g_S
        for (int lm = 0; lm < 16; ++lm) {
            int l = kLML[lm];
            launch_gemm<false,false,false,false,false>(
                stream, gAbuf + lm * CC, 16 * CC, Wposto[i] + (size_t)l * CC * CC, CC,
                Araw + lm * CC, 16 * CC, nullptr, 0, NPAD, CC, CC);
        }
        hipMemsetAsync(ghu, 0, (size_t)NPAD * CC * 4, stream);
        mace_msg_bwd_k<<<EE, 128, 0, stream>>>(wbuf, Ybuf, hu, eidx, Araw, gY, ghu);
        // MLP backward: g_z3..g_z1, then g_rf += g_z1 @ W1^T
        launch_gemm<true,false,true,false,true>(stream, wbuf, 512, W4o[i], 512,
                                                gza, 64, z3, 64, EE, 64, 512);
        launch_gemm<true,false,true,false,true>(stream, gza, 64, W3o[i], 64,
                                                gzb, 64, z2, 64, EE, 64, 64);
        launch_gemm<true,false,true,false,true>(stream, gzb, 64, W2o[i], 64,
                                                gza, 64, z1, 64, EE, 64, 64);
        launch_gemm<true,false,false,true,false>(stream, gza, 64, W1o[i], 64,
                                                 grf, 8, nullptr, 0, EE, 8, 64);
        // g_h += g_hu @ Wup^T
        launch_gemm<false,false,false,true,false>(stream, ghu, CC, Wupo[i], CC,
                                                  g_h, CC, nullptr, 0, NPAD, CC, CC);
        if (i == 1)  // seed layer-0 upstream grad: dE/dh1 = g_h + w_r0
            mace_add_r0_k<<<NN, 128, 0, stream>>>(g_h, wr0, ghcur);
    }
    mace_geom_bwd_k<<<(EE + 255) / 256, 256, 0, stream>>>(
        vecb, rbuf, ribuf, gY, grf, eidx, forces);
}
